// ChannelWiseRNNClassifier_84911503441936
// MI455X (gfx1250) — matrix operand order repaired and checked by reading, not compiled, not run on hardware
//
#include <hip/hip_runtime.h>
#include <hip/hip_bf16.h>

// CDNA5 / gfx1250, wave32. WMMA f32 = f16 x f16 + f32.
typedef __attribute__((ext_vector_type(16))) _Float16 v16h;
typedef __attribute__((ext_vector_type(8)))  float    v8f;

union V16H {
    v16h     v;
    uint4    q[2];
    _Float16 h[16];
};

// ---- problem constants ----
#define GB   256   // batch
#define GC   16    // channels
#define GI   64    // input features
#define GH   40    // hidden
#define GT   512   // timesteps
#define G3H  120   // 3*H
#define TT   16    // time chunk staged in LDS
#define NCK  (GT / TT)   // 32 chunks
#define XS   72    // padded i-stride of x/h f16 LDS rows (bank spread: 72*2B=144B)
#define XROW (16*XS)     // one timestep row of x16: 1152 f16
#define GS   132   // padded gate-row stride (floats); 8*132 % 64banks == 32 -> no alias

// Workgroup barrier that orders ONLY LDS traffic (s_wait_dscnt), leaving global
// prefetch loads (LOADcnt) in flight. __syncthreads() would drain LOADcnt too.
__device__ inline void lds_barrier() {
    asm volatile("s_wait_dscnt 0x0\n\t"
                 "s_barrier_signal -1\n\t"
                 "s_barrier_wait -1" ::: "memory");
}

// A-matrix gather (16-bit A 16x32 ISA layout) from row-major [m][stride] f16 LDS.
// lane<16: K = {k0..k0+7, k0+16..k0+23} with k0=0 ; lane>=16: k0=8.
__device__ inline v16h load_A_lds(const _Float16* base, int m, int stride_h,
                                  int kbase, int k0) {
    V16H r;
    const _Float16* p = base + m * stride_h + kbase + k0;
    r.q[0] = *(const uint4*)(p);        // K = kbase+k0 .. +7
    r.q[1] = *(const uint4*)(p + 16);   // K = kbase+k0+16 .. +23
    return r.v;
}

// B-matrix tile (32x16, 16-bit ISA layout) built from global f32 row-major W[g][k].
// lane holds column n=g ; K = {kbase..kbase+7, kbase+16..kbase+23} (kbase includes 8*(lane>=16)).
// Zero-pads K >= kdim and rows g >= grows. Init-time only.
__device__ inline v16h load_B_gbl(const float* W, int g, int kdim, int kbase, int grows) {
    V16H r;
#pragma unroll
    for (int e = 0; e < 16; ++e) {
        int k = kbase + (e < 8 ? e : e + 8);
        float v = (g < grows && k < kdim) ? W[g * kdim + k] : 0.0f;
        r.h[e] = (_Float16)v;
    }
    return r.v;
}

// Fast activations: v_exp_f32 + v_rcp_f32, no IEEE division sequences.
__device__ inline float fsigmoid(float x) {
    return __builtin_amdgcn_rcpf(1.0f + __expf(-x));
}
__device__ inline float ftanh(float x) {
    // tanh(x) = 1 - 2/(exp(2x)+1); saturates correctly at +/-1.
    return 1.0f - 2.0f * __builtin_amdgcn_rcpf(__expf(2.0f * x) + 1.0f);
}

__global__ __launch_bounds__(256)
void ChannelWiseRNNClassifier_gru_kernel(const float* __restrict__ x,
                                         const float* __restrict__ W_ih,
                                         const float* __restrict__ W_hh,
                                         const float* __restrict__ b_ih,
                                         const float* __restrict__ b_hh,
                                         const float* __restrict__ W_fc,
                                         const float* __restrict__ b_fc,
                                         float* __restrict__ out) {
    __shared__ _Float16 x16[TT * XROW];     // [t][b][i] f16, i-stride XS      36864 B
    __shared__ _Float16 h16[16 * XS];       // [b][k] f16, K padded 40->64      2304 B
    __shared__ float    h32[16 * GH];       // hidden, written only at the end  2560 B
    __shared__ float    gi_s[16 * GS];      // input-proj gates (b x 128pad)    8448 B
    __shared__ float    gh_s[16 * GS];      // recurrent gates                  8448 B

    const int c    = blockIdx.x & 15;       // channel
    const int b0   = (blockIdx.x >> 4) * 16;// batch tile base
    const int tid  = threadIdx.x;
    const int w    = tid >> 5;              // wave id 0..7 (owns gate cols w*16..+15)
    const int lane = tid & 31;
    const int lrow = lane & 15;
    const int hi   = lane >> 4;             // lane group
    const int k0   = hi * 8;                // A-gather K offset
    const int kbB  = hi * 8;                // B-gather K offset

    // ---- zero hidden state (h16 K-pad region stays zero forever) ----
    for (int idx = tid; idx < 16 * XS; idx += 256) h16[idx] = (_Float16)0.0f;

    // ---- resident B tiles (weights), zero-padded ----
    const float* Wi = W_ih + (size_t)c * G3H * GI;
    const float* Wh = W_hh + (size_t)c * G3H * GH;
    const int g = w * 16 + lrow;
    const v16h Bi0 = load_B_gbl(Wi, g, GI, kbB,      G3H);
    const v16h Bi1 = load_B_gbl(Wi, g, GI, kbB + 32, G3H);
    const v16h Bh0 = load_B_gbl(Wh, g, GH, kbB,      G3H);
    const v16h Bh1 = load_B_gbl(Wh, g, GH, kbB + 32, G3H);

    const int nc = w * 16 + lrow;           // gate column written by this lane

    // ---- per-thread elementwise cells: tid, tid+256 (always valid), tid+512 (tid<128) ----
    const bool ev2 = (tid < 128);
    int   eb[3], ek[3], og[3], oh16v[3];
    float bir[3], biz[3], bin_[3], bhr[3], bhz[3], bhn[3];
    float hreg[3];
#pragma unroll
    for (int j = 0; j < 3; ++j) {
        const int cell = tid + 256 * j;
        const int b = cell / GH;
        const int k = cell - b * GH;
        eb[j] = b; ek[j] = k;
        og[j]    = b * GS + k;      // same offset valid for gi_s and gh_s
        oh16v[j] = b * XS + k;
        hreg[j]  = 0.0f;
        if (j < 2 || ev2) {
            bir[j]  = b_ih[c * G3H + k];
            biz[j]  = b_ih[c * G3H + 40 + k];
            bin_[j] = b_ih[c * G3H + 80 + k];
            bhr[j]  = b_hh[c * G3H + k];
            bhz[j]  = b_hh[c * G3H + 40 + k];
            bhn[j]  = b_hh[c * G3H + 80 + k];
        } else {
            bir[j] = biz[j] = bin_[j] = bhr[j] = bhz[j] = bhn[j] = 0.0f;
        }
    }

    // ---- x staging geometry: this thread owns (i = tid>>2, tc = (tid&3)*4),
    //      and loads batch row m = it for it = 0..15. ----
    const int rbase = tid >> 2;             // input feature i (0..63)
    const int tc    = (tid & 3) * 4;        // t offset within chunk
    const float* xbase = x + ((size_t)(b0 * GC + c) * GI) * GT
                           + (size_t)rbase * GT + tc;
    const size_t xmstride = (size_t)GC * GI * GT;   // stride between batch rows

    // ---- prologue: issue chunk 0 loads ----
    float4 vx[16];
#pragma unroll
    for (int it = 0; it < 16; ++it)
        vx[it] = *(const float4*)(xbase + (size_t)it * xmstride);

    for (int ck = 0; ck < NCK; ++ck) {
        // ---- convert & store staged chunk (waits on outstanding loads) ----
#pragma unroll
        for (int it = 0; it < 16; ++it) {
            const int base = it * XS + rbase;
            x16[(tc + 0) * XROW + base] = (_Float16)vx[it].x;
            x16[(tc + 1) * XROW + base] = (_Float16)vx[it].y;
            x16[(tc + 2) * XROW + base] = (_Float16)vx[it].z;
            x16[(tc + 3) * XROW + base] = (_Float16)vx[it].w;
        }
        lds_barrier();

        // ---- issue next chunk's global loads; they retire during the 16 steps ----
        if (ck + 1 < NCK) {
            const float* xnext = xbase + (ck + 1) * TT;
#pragma unroll
            for (int it = 0; it < 16; ++it)
                vx[it] = *(const float4*)(xnext + (size_t)it * xmstride);
        }

        for (int t = 0; t < TT; ++t) {
            // ---- issue all A-tile gathers, then 4 WMMAs ----
            const _Float16* xt = &x16[t * XROW];
            const v16h xa0 = load_A_lds(xt,  lrow, XS, 0,  k0);
            const v16h xa1 = load_A_lds(xt,  lrow, XS, 32, k0);
            const v16h ha0 = load_A_lds(h16, lrow, XS, 0,  k0);
            const v16h ha1 = load_A_lds(h16, lrow, XS, 32, k0);

            v8f acc_i = {};
            acc_i = __builtin_amdgcn_wmma_f32_16x16x32_f16(false, xa0, false, Bi0,
                                                           (short)0, acc_i, false, false);
            acc_i = __builtin_amdgcn_wmma_f32_16x16x32_f16(false, xa1, false, Bi1,
                                                           (short)0, acc_i, false, false);
            v8f acc_h = {};
            acc_h = __builtin_amdgcn_wmma_f32_16x16x32_f16(false, ha0, false, Bh0,
                                                           (short)0, acc_h, false, false);
            acc_h = __builtin_amdgcn_wmma_f32_16x16x32_f16(false, ha1, false, Bh1,
                                                           (short)0, acc_h, false, false);

            // ---- dump accumulators: VGPR j -> (M = j + 8*hi, N = nc) ----
#pragma unroll
            for (int j = 0; j < 8; ++j) {
                const int m = j + hi * 8;
                gi_s[m * GS + nc] = acc_i[j];
                gh_s[m * GS + nc] = acc_h[j];
            }
            lds_barrier();

            // ---- GRU cell update: cells 0,1 unconditional, cell 2 only for tid<128 ----
#pragma unroll
            for (int j = 0; j < 3; ++j) {
                if (j < 2 || ev2) {
                    const float gir = gi_s[og[j]];
                    const float giz = gi_s[og[j] + 40];
                    const float gin = gi_s[og[j] + 80];
                    const float ghr = gh_s[og[j]];
                    const float ghz = gh_s[og[j] + 40];
                    const float ghn = gh_s[og[j] + 80];
                    const float rg = fsigmoid(gir + bir[j] + ghr + bhr[j]);
                    const float zg = fsigmoid(giz + biz[j] + ghz + bhz[j]);
                    const float ng = ftanh(gin + bin_[j] + rg * (ghn + bhn[j]));
                    const float hn = (1.0f - zg) * ng + zg * hreg[j];
                    hreg[j] = hn;
                    h16[oh16v[j]] = (_Float16)hn;
                }
            }
            lds_barrier();
        }
    }

    // ---- spill hidden state once, then channel-wise FC (H -> 1) + sigmoid ----
#pragma unroll
    for (int j = 0; j < 3; ++j)
        if (j < 2 || ev2) h32[eb[j] * GH + ek[j]] = hreg[j];
    lds_barrier();

    if (tid < 16) {
        float acc = b_fc[c];
        const float* wf = W_fc + c * GH;
#pragma unroll
        for (int k = 0; k < GH; ++k) acc += h32[tid * GH + k] * wf[k];
        out[(size_t)(b0 + tid) * GC + c] = fsigmoid(acc);
    }
}

extern "C" void kernel_launch(void* const* d_in, const int* in_sizes, int n_in,
                              void* d_out, int out_size, void* d_ws, size_t ws_size,
                              hipStream_t stream) {
    const float* x    = (const float*)d_in[0];
    const float* W_ih = (const float*)d_in[1];
    const float* W_hh = (const float*)d_in[2];
    const float* b_ih = (const float*)d_in[3];
    const float* b_hh = (const float*)d_in[4];
    const float* W_fc = (const float*)d_in[5];
    const float* b_fc = (const float*)d_in[6];
    float* out = (float*)d_out;

    dim3 grid(GC * (GB / 16));   // 256 workgroups: (channel, batch-tile)
    dim3 block(256);             // 8 wave32 waves
    ChannelWiseRNNClassifier_gru_kernel<<<grid, block, 0, stream>>>(
        x, W_ih, W_hh, b_ih, b_hh, W_fc, b_fc, out);
}
